// LinearAttentionLayer_69939247448427
// MI455X (gfx1250) — matrix operand order repaired and checked by reading, not compiled, not run on hardware
//
#include <hip/hip_runtime.h>
#include <cstddef>
#include <cstdint>

typedef __attribute__((ext_vector_type(16))) _Float16 v16h;
typedef __attribute__((ext_vector_type(8)))  _Float16 v8h;
typedef __attribute__((ext_vector_type(4)))  _Float16 v4h;
typedef __attribute__((ext_vector_type(8)))  float    v8f;

// ---------------------------------------------------------------------------
// Problem constants (from the reference)
// ---------------------------------------------------------------------------
constexpr int NB = 4;        // batch
constexpr int LL = 4096;     // query length
constexpr int SS = 4096;     // key length
constexpr int DM = 1024;     // d_model
constexpr int NH = 16;       // heads
constexpr int DH = 64;       // head dim

// ---------------------------------------------------------------------------
// CDNA5 async copy: global -> LDS, 16 bytes per lane.
//   vdst  = LDS byte offset (VGPR), GVS addressing: SADDR(u64) + VADDR(u32)
// LDS generic pointers: aperture form, addr[31:0] == LDS byte offset.
// ---------------------------------------------------------------------------
__device__ __forceinline__ void async_copy_b128(uint32_t lds_byte,
                                                const void* gbase,
                                                uint32_t goff_byte) {
  asm volatile("global_load_async_to_lds_b128 %0, %1, %2"
               :: "v"(lds_byte), "v"(goff_byte), "s"(gbase)
               : "memory");
}

__device__ __forceinline__ void wait_async0() {
#if __has_builtin(__builtin_amdgcn_s_wait_asynccnt)
  __builtin_amdgcn_s_wait_asynccnt(0);
#else
  asm volatile("s_wait_asynccnt 0x0" ::: "memory");
#endif
}

// ---------------------------------------------------------------------------
// WMMA fragment helpers.
//   A-frag: element i -> K = kb + (i<8 ? i : i+8)   (hiOff = 16)
//   B-frag: element i -> K = kb + (i<8 ? i : i+8)   (hiOff = 16)
// ---------------------------------------------------------------------------
__device__ __forceinline__ v16h make_frag(const _Float16* __restrict__ base,
                                          int hiOffElems) {
  v8h lo = *(const v8h*)(base);
  v8h hi = *(const v8h*)(base + hiOffElems);
  v16h r;
#pragma unroll
  for (int i = 0; i < 8; ++i) { r[i] = lo[i]; r[i + 8] = hi[i]; }
  return r;
}

// ---------------------------------------------------------------------------
// Kernel: elementwise fp32 -> f16 (vectorized x4)
// ---------------------------------------------------------------------------
__global__ __launch_bounds__(256)
void cvt_f16_kernel(const float* __restrict__ in, _Float16* __restrict__ out,
                    size_t n4) {
  size_t i = (size_t)blockIdx.x * blockDim.x + threadIdx.x;
  if (i >= n4) return;
  float4 t = ((const float4*)in)[i];
  v4h o;
  o[0] = (_Float16)t.x; o[1] = (_Float16)t.y;
  o[2] = (_Float16)t.z; o[3] = (_Float16)t.w;
  ((v4h*)out)[i] = o;
}

// ---------------------------------------------------------------------------
// Kernel: W[K=1024][N=1024] fp32 -> WT[N][K] f16 (LDS-tiled transpose)
// ---------------------------------------------------------------------------
__global__ __launch_bounds__(256)
void wtrans_kernel(const float* __restrict__ W, _Float16* __restrict__ WT) {
  __shared__ float t[32][33];
  const int bx = blockIdx.x * 32, by = blockIdx.y * 32;
  const int tx = threadIdx.x & 31, ty = threadIdx.x >> 5;  // 32 x 8
#pragma unroll
  for (int j = 0; j < 4; ++j)
    t[ty + j * 8][tx] = W[(size_t)(by + ty + j * 8) * DM + bx + tx];
  __syncthreads();
#pragma unroll
  for (int j = 0; j < 4; ++j)
    WT[(size_t)(bx + ty + j * 8) * DM + by + tx] = (_Float16)t[tx][ty + j * 8];
}

// ---------------------------------------------------------------------------
// Main GEMM: Out[M,N] = epilogue(A[M,K](f16) @ BT[N,K](f16)^T + bias)
//   Block tile 128x128, BK=64, double-buffered LDS filled by
//   global_load_async_to_lds_b128; 8 waves, wave tile 64x32 (4x2 acc),
//   16 v_wmma_f32_16x16x32_f16 per K-step per wave.
// ---------------------------------------------------------------------------
__device__ __forceinline__ void issue_tile_async(const _Float16* __restrict__ g,
                                                 int ldK, int rowBase, int k0,
                                                 _Float16* ldsBase, int tid) {
  // 128 rows x 64 f16 = 1024 16B chunks; 4 chunks per thread
#pragma unroll
  for (int j = 0; j < 4; ++j) {
    const int ch  = tid * 4 + j;
    const int row = ch >> 3;
    const int cb  = (ch & 7) * 16;  // byte offset within row
    const uint32_t lds = (uint32_t)(uintptr_t)(ldsBase + row * 72) + cb;
    const uint32_t gof = (uint32_t)(((size_t)(rowBase + row) * ldK + k0) * 2) + cb;
    async_copy_b128(lds, g, gof);
  }
}

template <int ACT, typename OType>
__global__ __launch_bounds__(256)
void gemm_async_wmma_kernel(const _Float16* __restrict__ A,
                            const _Float16* __restrict__ BT,
                            const float* __restrict__ bias,
                            OType* __restrict__ Out, int M, int N, int K) {
  constexpr int BM = 128, BN = 128, BK = 64, LD = 72;  // 144B row stride (16B-mult)
  __shared__ _Float16 ldsA[2][BM * LD];
  __shared__ _Float16 ldsB[2][BN * LD];
  const int tid  = threadIdx.x;
  const int wave = tid >> 5;
  const int lane = tid & 31;
  const int bm = blockIdx.x * BM;
  const int bn = blockIdx.y * BN;
  const int wm = (wave >> 2) * 64;
  const int wn = (wave & 3) * 32;
  const int r  = lane & 15;
  const int hi = lane >> 4;

  v8f acc[4][2];
#pragma unroll
  for (int i = 0; i < 4; ++i)
#pragma unroll
    for (int j = 0; j < 2; ++j) acc[i][j] = v8f{};

  const int nk = K / BK;
  issue_tile_async(A,  K, bm, 0, &ldsA[0][0], tid);
  issue_tile_async(BT, K, bn, 0, &ldsB[0][0], tid);

  int cur = 0;
  for (int t = 0; t < nk; ++t) {
    wait_async0();        // this wave's async copies for buffer `cur` done
    __syncthreads();      // ... for all waves
    if (t + 1 < nk) {
      issue_tile_async(A,  K, bm, (t + 1) * BK, &ldsA[cur ^ 1][0], tid);
      issue_tile_async(BT, K, bn, (t + 1) * BK, &ldsB[cur ^ 1][0], tid);
    }
#pragma unroll
    for (int ks = 0; ks < BK; ks += 32) {
      v16h af[4], bf[2];
#pragma unroll
      for (int mt = 0; mt < 4; ++mt)
        af[mt] = make_frag(&ldsA[cur][(wm + mt * 16 + r) * LD + ks + hi * 8], 16);
#pragma unroll
      for (int nt = 0; nt < 2; ++nt)
        bf[nt] = make_frag(&ldsB[cur][(wn + nt * 16 + r) * LD + ks + hi * 8], 16);
#pragma unroll
      for (int mt = 0; mt < 4; ++mt)
#pragma unroll
        for (int nt = 0; nt < 2; ++nt)
          acc[mt][nt] = __builtin_amdgcn_wmma_f32_16x16x32_f16(
              false, af[mt], false, bf[nt], (short)0, acc[mt][nt], false, false);
    }
    __syncthreads();      // all waves done reading `cur` before it is refilled
    cur ^= 1;
  }

#pragma unroll
  for (int mt = 0; mt < 4; ++mt) {
#pragma unroll
    for (int nt = 0; nt < 2; ++nt) {
      const int col = bn + wn + nt * 16 + r;
      const float bv = bias[col];
#pragma unroll
      for (int v = 0; v < 8; ++v) {
        const int row = bm + wm + mt * 16 + hi * 8 + v;
        float x = acc[mt][nt][v] + bv;
        if (ACT) x = (x > 0.f) ? (x + 1.f) : __expf(x);  // elu(x)+1
        Out[(size_t)row * N + col] = (OType)x;
      }
    }
  }
}

// ---------------------------------------------------------------------------
// K_sum[n, c] = sum_s K_lin_f16[n, s, c]
// ---------------------------------------------------------------------------
__global__ __launch_bounds__(256)
void ksum_kernel(const _Float16* __restrict__ Kf, float* __restrict__ Ksum, int S) {
  const int idx = blockIdx.x * blockDim.x + threadIdx.x;  // 0..4095
  const int n = idx >> 10, c = idx & 1023;
  const _Float16* p = Kf + (size_t)n * S * DM + c;
  float s = 0.f;
  for (int i = 0; i < S; ++i) s += (float)p[(size_t)i * DM];
  Ksum[idx] = s;
}

// ---------------------------------------------------------------------------
// KV[n,h,m,d] = sum_s V_f16[n,s,h,m] * K_lin_f16[n,s,h,d]
//   Block = (n,h); 16 waves, each owns one 16x16 (m,d) tile over all S.
// ---------------------------------------------------------------------------
__global__ __launch_bounds__(512)
void kv_wmma_kernel(const _Float16* __restrict__ Kf, const _Float16* __restrict__ Vf,
                    float* __restrict__ KV, int S) {
  constexpr int LDT = 40;
  __shared__ _Float16 ldsV[DH * LDT];  // [m][s] (K-fastest)
  __shared__ _Float16 ldsK[DH * LDT];  // [d][s]
  const int n = blockIdx.x, h = blockIdx.y;
  const int tid = threadIdx.x, wave = tid >> 5, lane = tid & 31;
  const int mt = wave >> 2, dt = wave & 3;
  const int r = lane & 15, hi = lane >> 4;
  const _Float16* Kb = Kf + (size_t)n * S * DM + h * DH;
  const _Float16* Vb = Vf + (size_t)n * S * DM + h * DH;

  v8f acc = v8f{};
  for (int s0 = 0; s0 < S; s0 += 32) {
    __syncthreads();
    {
      const int srow = tid >> 4;
      const int c0   = (tid & 15) * 4;
      v4h k4 = *(const v4h*)(Kb + (size_t)(s0 + srow) * DM + c0);
      v4h v4 = *(const v4h*)(Vb + (size_t)(s0 + srow) * DM + c0);
#pragma unroll
      for (int i = 0; i < 4; ++i) {
        ldsK[(c0 + i) * LDT + srow] = k4[i];
        ldsV[(c0 + i) * LDT + srow] = v4[i];
      }
    }
    __syncthreads();
    v16h a = make_frag(&ldsV[(mt * 16 + r) * LDT + hi * 8], 16);   // A[m,s]=V[s,m]
    v16h b = make_frag(&ldsK[(dt * 16 + r) * LDT + hi * 8], 16);   // B[s,d]=K[s,d]
    acc = __builtin_amdgcn_wmma_f32_16x16x32_f16(false, a, false, b,
                                                 (short)0, acc, false, false);
  }
  float* out = KV + (((size_t)n * NH + h) * DH) * DH;
#pragma unroll
  for (int v = 0; v < 8; ++v)
    out[(mt * 16 + hi * 8 + v) * DH + dt * 16 + r] = acc[v];
}

// ---------------------------------------------------------------------------
// V_lin[n,l,h,m] = (Q_lin . KV[m,:]) / (Q_lin . K_sum + eps)
//   Normalizer folded in as B column 64 (B widened to 80 cols).
// ---------------------------------------------------------------------------
__global__ __launch_bounds__(256)
void apply_wmma_kernel(const _Float16* __restrict__ Qf, const float* __restrict__ KV,
                       const float* __restrict__ Ksum, _Float16* __restrict__ Vlin,
                       int L) {
  constexpr int LDB = 72;
  __shared__ _Float16 ldsB[80 * LDB];
  __shared__ float ldsDen[128];
  const int n = blockIdx.z, h = blockIdx.y;
  const int l0 = blockIdx.x * 128;
  const int tid = threadIdx.x, wave = tid >> 5, lane = tid & 31;
  const int r = lane & 15, hi = lane >> 4;

  {  // cols 0..63: ldsB[m][d] = KV[m][d]
    const float* kv = KV + (((size_t)n * NH + h) * DH) * DH;
    const int m = tid >> 2;
    const int d0 = (tid & 3) * 16;
    const float4* s4 = (const float4*)(kv + m * DH + d0);
#pragma unroll
    for (int j = 0; j < 4; ++j) {
      float4 t = s4[j];
      ldsB[m * LDB + d0 + j * 4 + 0] = (_Float16)t.x;
      ldsB[m * LDB + d0 + j * 4 + 1] = (_Float16)t.y;
      ldsB[m * LDB + d0 + j * 4 + 2] = (_Float16)t.z;
      ldsB[m * LDB + d0 + j * 4 + 3] = (_Float16)t.w;
    }
  }
  {  // col 64 = K_sum; cols 65..79 = 0
    const float* ks = Ksum + ((size_t)n * NH + h) * DH;
    const int c = 64 + (tid >> 4);
    const int d0 = (tid & 15) * 4;
#pragma unroll
    for (int i = 0; i < 4; ++i)
      ldsB[c * LDB + d0 + i] = (c == 64) ? (_Float16)ks[d0 + i] : (_Float16)0.f;
  }
  __syncthreads();

  const _Float16* Qrow = Qf + ((size_t)n * L + l0 + wave * 16) * DM + h * DH;
  v8f acc[5];
#pragma unroll
  for (int i = 0; i < 5; ++i) acc[i] = v8f{};

#pragma unroll
  for (int k0 = 0; k0 < DH; k0 += 32) {
    v16h a = make_frag(Qrow + (size_t)r * DM + k0 + hi * 8, 16);
#pragma unroll
    for (int nt = 0; nt < 5; ++nt) {
      v16h b = make_frag(&ldsB[(nt * 16 + r) * LDB + k0 + hi * 8], 16);
      acc[nt] = __builtin_amdgcn_wmma_f32_16x16x32_f16(false, a, false, b,
                                                       (short)0, acc[nt], false, false);
    }
  }

  if (r == 0) {
#pragma unroll
    for (int v = 0; v < 8; ++v) ldsDen[wave * 16 + hi * 8 + v] = acc[4][v];
  }
  __syncthreads();

  _Float16* out = Vlin + ((size_t)n * L + l0 + wave * 16) * DM + h * DH;
#pragma unroll
  for (int mt = 0; mt < 4; ++mt) {
#pragma unroll
    for (int v = 0; v < 8; ++v) {
      const int row = hi * 8 + v;
      const float den = ldsDen[wave * 16 + row] + 1e-10f;
      out[(size_t)row * DM + mt * 16 + r] = (_Float16)(acc[mt][v] / den);
    }
  }
}

// ---------------------------------------------------------------------------
// Host launcher
// ---------------------------------------------------------------------------
extern "C" void kernel_launch(void* const* d_in, const int* in_sizes, int n_in,
                              void* d_out, int out_size, void* d_ws, size_t ws_size,
                              hipStream_t stream) {
  (void)in_sizes; (void)n_in; (void)out_size; (void)ws_size;
  const float* queries = (const float*)d_in[0];
  const float* keys    = (const float*)d_in[1];
  const float* values  = (const float*)d_in[2];
  const float* Wq = (const float*)d_in[3];  const float* bq = (const float*)d_in[4];
  const float* Wk = (const float*)d_in[5];  const float* bk = (const float*)d_in[6];
  const float* Wv = (const float*)d_in[7];  const float* bv = (const float*)d_in[8];
  const float* Wo = (const float*)d_in[9];  const float* bo = (const float*)d_in[10];
  float* out = (float*)d_out;

  const size_t ROWS = (size_t)NB * LL;  // 16384
  const size_t E = ROWS * DM;           // elems per big tensor
  char* ws = (char*)d_ws;
  _Float16* Xq   = (_Float16*)ws;       // reused as Vlin after Q projection
  _Float16* Xk   = Xq + E;
  _Float16* Xv   = Xk + E;
  _Float16* Qlin = Xv + E;
  _Float16* Klin = Qlin + E;
  _Float16* Vf   = Klin + E;
  _Float16* WqT  = Vf + E;
  _Float16* WkT  = WqT + (size_t)DM * DM;
  _Float16* WvT  = WkT + (size_t)DM * DM;
  _Float16* WoT  = WvT + (size_t)DM * DM;
  float*    KV   = (float*)(WoT + (size_t)DM * DM);
  float*    Ksum = KV + (size_t)NB * NH * DH * DH;
  _Float16* Vlin = Xq;

  // Stage 0: precision staging (one-time f16 conversion + weight transpose)
  const size_t n4 = E / 4;
  cvt_f16_kernel<<<(unsigned)(n4 / 256), 256, 0, stream>>>(queries, Xq, n4);
  cvt_f16_kernel<<<(unsigned)(n4 / 256), 256, 0, stream>>>(keys,    Xk, n4);
  cvt_f16_kernel<<<(unsigned)(n4 / 256), 256, 0, stream>>>(values,  Xv, n4);
  wtrans_kernel<<<dim3(32, 32), 256, 0, stream>>>(Wq, WqT);
  wtrans_kernel<<<dim3(32, 32), 256, 0, stream>>>(Wk, WkT);
  wtrans_kernel<<<dim3(32, 32), 256, 0, stream>>>(Wv, WvT);
  wtrans_kernel<<<dim3(32, 32), 256, 0, stream>>>(Wo, WoT);

  // Stage 1: projections (async double-buffered WMMA GEMMs)
  const dim3 gProj(128, 8);
  gemm_async_wmma_kernel<1, _Float16><<<gProj, 256, 0, stream>>>(
      Xq, WqT, bq, Qlin, (int)ROWS, DM, DM);
  gemm_async_wmma_kernel<1, _Float16><<<gProj, 256, 0, stream>>>(
      Xk, WkT, bk, Klin, (int)ROWS, DM, DM);
  gemm_async_wmma_kernel<0, _Float16><<<gProj, 256, 0, stream>>>(
      Xv, WvT, bv, Vf, (int)ROWS, DM, DM);

  // Stage 2: K_sum and KV accumulation
  ksum_kernel<<<16, 256, 0, stream>>>(Klin, Ksum, SS);
  kv_wmma_kernel<<<dim3(NB, NH), 512, 0, stream>>>(Klin, Vf, KV, SS);

  // Stage 3: normalized attention output (f16)
  apply_wmma_kernel<<<dim3(LL / 128, NH, NB), 256, 0, stream>>>(
      Qlin, KV, Ksum, Vlin, LL);

  // Stage 4: final projection (f32 out)
  gemm_async_wmma_kernel<0, float><<<gProj, 256, 0, stream>>>(
      Vlin, WoT, bo, out, (int)ROWS, DM, DM);
}
